// TransitionUp_32169305047426
// MI455X (gfx1250) — matrix operand order repaired and checked by reading, not compiled, not run on hardware
//
#include <hip/hip_runtime.h>
#include <cstdint>
#include <cstddef>

#define N_PTS 24576
#define M_SUB 6144
#define CIN   512
#define COUT  256

typedef __attribute__((ext_vector_type(16))) _Float16 v16h;
typedef __attribute__((ext_vector_type(8)))  float    v8f;

struct B32x8 { uint4 a, b; };
static __device__ __forceinline__ v16h load_v16h(const _Float16* p) {
    return __builtin_bit_cast(v16h, *reinterpret_cast<const B32x8*>(p));
}

// ---------------------------------------------------------------------------
// Pack row-major f32 A[M x K] into a wave32 WMMA A layout (contiguous k order, same as packed B):
// dst[((mb*KB + kb)*32 + lane)*16 + e] = (f16) A[m, k]
//   m = mb*16 + lane%16
//   k = kb*32 + (lane>=16 ? 16 : 0) + e
// Each lane's 16 halfs are contiguous (32B) -> 2x global_load_b128 in the GEMM.
// ---------------------------------------------------------------------------
__global__ void pack_a_kernel(const float* __restrict__ A, _Float16* __restrict__ out,
                              int Kcols) {
    int tid = blockIdx.x * blockDim.x + threadIdx.x;
    int KB  = Kcols >> 5;
    int e    = tid & 15;
    int lane = (tid >> 4) & 31;
    int kb   = (tid >> 9) % KB;
    int mb   = (tid >> 9) / KB;
    int m = mb * 16 + (lane & 15);
    int k = kb * 32 + ((lane >> 4) << 4) + e;
    out[tid] = (_Float16)A[(size_t)m * Kcols + k];
}

// ---------------------------------------------------------------------------
// Pack row-major f32 B[K x N] into [k_blk][n][kk(0..31)] f16 so that lane l
// (column n = n0 + l%16, half = l>>4) reads 16 contiguous halfs at kk = half*16.
// ---------------------------------------------------------------------------
__global__ void pack_b_kernel(const float* __restrict__ B, _Float16* __restrict__ out,
                              int Ncols) {
    int tid = blockIdx.x * blockDim.x + threadIdx.x;
    int n = tid % Ncols;
    int k = tid / Ncols;
    out[(size_t)(k >> 5) * Ncols * 32 + (size_t)n * 32 + (k & 31)] = (_Float16)B[tid];
}

// ---------------------------------------------------------------------------
// GEMM: C[Mrows x Ncols] = A * B + bias, f16 inputs (packed), f32 accumulate.
// One wave computes a 32x32 output tile: 4 WMMA accumulators, A-pair x B-pair.
// Steady state: 8x global_load_b128 : 4x v_wmma (2:1), 4 independent WMMAs
// per k-step for hazard-free interleaving.
// ---------------------------------------------------------------------------
__global__ void gemm_f16_kernel(const _Float16* __restrict__ Apk,
                                const _Float16* __restrict__ Bpk,
                                const float* __restrict__ bias,
                                float* __restrict__ C,
                                int tiles_m32, int tiles_n32, int KB, int Ncols) {
    int wave = (int)((blockIdx.x * blockDim.x + threadIdx.x) >> 5);
    int lane = threadIdx.x & 31;
    if (wave >= tiles_m32 * tiles_n32) return;
    int tm = wave / tiles_n32;        // 32-row tile index
    int n0 = (wave % tiles_n32) * 32; // 32-col tile base

    const _Float16* ap0 = Apk + (size_t)(2 * tm) * KB * 512 + lane * 16;
    const _Float16* ap1 = ap0 + (size_t)KB * 512;       // next 16-row block
    int col = n0 + (lane & 15);
    const _Float16* bp = Bpk + (size_t)col * 32 + (lane >> 4) * 16;
    size_t bstride = (size_t)Ncols * 32;

    v8f acc00 = {}; v8f acc01 = {}; v8f acc10 = {}; v8f acc11 = {};
    for (int kb = 0; kb < KB; ++kb) {
        v16h a0 = load_v16h(ap0); ap0 += 512;
        v16h a1 = load_v16h(ap1); ap1 += 512;
        v16h b0 = load_v16h(bp);
        v16h b1 = load_v16h(bp + 512);   // next 16 columns
        bp += bstride;
        acc00 = __builtin_amdgcn_wmma_f32_16x16x32_f16(false, a0, false, b0,
                                                       (short)0, acc00, false, false);
        acc01 = __builtin_amdgcn_wmma_f32_16x16x32_f16(false, a0, false, b1,
                                                       (short)0, acc01, false, false);
        acc10 = __builtin_amdgcn_wmma_f32_16x16x32_f16(false, a1, false, b0,
                                                       (short)0, acc10, false, false);
        acc11 = __builtin_amdgcn_wmma_f32_16x16x32_f16(false, a1, false, b1,
                                                       (short)0, acc11, false, false);
    }
    // D layout: element v -> row = v + (lane<16 ? 0 : 8), col = lane%16
    float bi0 = bias[col], bi1 = bias[col + 16];
    int r0 = tm * 32 + ((lane >> 4) << 3);
    float* cp0 = C + (size_t)r0 * Ncols + col;
    float* cp1 = cp0 + (size_t)16 * Ncols;
#pragma unroll
    for (int v = 0; v < 8; ++v) {
        cp0[0]  = acc00[v] + bi0;
        cp0[16] = acc01[v] + bi1;
        cp1[0]  = acc10[v] + bi0;
        cp1[16] = acc11[v] + bi1;
        cp0 += Ncols; cp1 += Ncols;
    }
}

// ---------------------------------------------------------------------------
// BatchNorm statistics: per-block partial (sum, sumsq) per channel (COUT=256).
// Deterministic two-pass reduction (no float atomics -> replay-stable).
// ---------------------------------------------------------------------------
__global__ void bn_partial_kernel(const float* __restrict__ H, float* __restrict__ part,
                                  int rows_per_block) {
    int c = threadIdx.x;
    const float* p = H + (size_t)blockIdx.x * rows_per_block * COUT + c;
    float s1 = 0.f, s2 = 0.f;
    for (int r = 0; r < rows_per_block; ++r) {
        float v = p[(size_t)r * COUT];
        s1 += v; s2 += v * v;
    }
    part[blockIdx.x * 512 + c]       = s1;
    part[blockIdx.x * 512 + 256 + c] = s2;
}

__global__ void bn_finalize_kernel(const float* __restrict__ part, int nblocks, float invM,
                                   const float* __restrict__ gamma,
                                   const float* __restrict__ beta,
                                   float* __restrict__ ss) {
    int c = threadIdx.x;
    float s1 = 0.f, s2 = 0.f;
    for (int b = 0; b < nblocks; ++b) {
        s1 += part[b * 512 + c];
        s2 += part[b * 512 + 256 + c];
    }
    float mu  = s1 * invM;
    float var = fmaxf(s2 * invM - mu * mu, 0.f);   // biased var (jnp.var)
    float sc  = gamma[c] * rsqrtf(var + 1e-5f);
    ss[c]       = sc;
    ss[256 + c] = beta[c] - mu * sc;
}

__global__ void bn_relu_kernel(float* __restrict__ H, const float* __restrict__ ss) {
    int i = blockIdx.x * blockDim.x + threadIdx.x;
    int c = i & (COUT - 1);
    H[i] = fmaxf(H[i] * ss[c] + ss[256 + c], 0.f);
}

// ---------------------------------------------------------------------------
// Brute-force 3-NN: entire pos_sub cloud lives in LDS as (x,y,z,|q|^2)
// (6144 float4 = 96 KB; CDNA5 WGP has 320 KB). Uniform loop index -> LDS
// broadcast reads (conflict-free). Ranking key s = |q|^2 - 2 p.q (order-
// equivalent to |p-q|^2 for fixed p) -> 3 FMAs per candidate. True squared
// distance recovered at the end as s + |p|^2 for the inverse-distance weights.
// ---------------------------------------------------------------------------
__global__ void knn_kernel(const float* __restrict__ pos, const float* __restrict__ pos_sub,
                           int* __restrict__ oidx, float* __restrict__ ow) {
    __shared__ float4 sp[M_SUB];
    for (int j = threadIdx.x; j < M_SUB; j += blockDim.x) {
        float qx = pos_sub[3 * j], qy = pos_sub[3 * j + 1], qz = pos_sub[3 * j + 2];
        sp[j] = make_float4(qx, qy, qz, fmaf(qx, qx, fmaf(qy, qy, qz * qz)));
    }
    __syncthreads();

    int n = blockIdx.x * blockDim.x + threadIdx.x;
    float px = pos[3 * n], py = pos[3 * n + 1], pz = pos[3 * n + 2];
    float pp = fmaf(px, px, fmaf(py, py, pz * pz));
    float ax = -2.f * px, ay = -2.f * py, az = -2.f * pz;
    float s0 = 3.4e38f, s1 = 3.4e38f, s2 = 3.4e38f;
    int   i0 = 0, i1 = 0, i2 = 0;
#pragma unroll 4
    for (int j = 0; j < M_SUB; ++j) {
        float4 q = sp[j];
        float s = fmaf(ax, q.x, fmaf(ay, q.y, fmaf(az, q.z, q.w)));
        if (s < s2) {
            if (s < s1) {
                if (s < s0) { s2 = s1; i2 = i1; s1 = s0; i1 = i0; s0 = s; i0 = j; }
                else        { s2 = s1; i2 = i1; s1 = s;  i1 = j; }
            } else          { s2 = s;  i2 = j; }
        }
    }
    float w0 = 1.f / fmaxf(s0 + pp, 1e-16f);
    float w1 = 1.f / fmaxf(s1 + pp, 1e-16f);
    float w2 = 1.f / fmaxf(s2 + pp, 1e-16f);
    float inv = 1.f / (w0 + w1 + w2);
    oidx[3 * n] = i0; oidx[3 * n + 1] = i1; oidx[3 * n + 2] = i2;
    ow[3 * n] = w0 * inv; ow[3 * n + 1] = w1 * inv; ow[3 * n + 2] = w2 * inv;
}

// ---------------------------------------------------------------------------
// Epilogue (in-place on d_out): out = ReLU(BN1(h1)) + sum_k w_k * f_sub[idx_k]
// One block per point, one thread per channel.
// ---------------------------------------------------------------------------
__global__ void final_kernel(float* __restrict__ out, const float* __restrict__ fsub,
                             const float* __restrict__ ss,
                             const int* __restrict__ idx, const float* __restrict__ w) {
    int n = blockIdx.x, c = threadIdx.x;
    int   i0 = idx[3 * n],  i1 = idx[3 * n + 1], i2 = idx[3 * n + 2];
    float w0 = w[3 * n],    w1 = w[3 * n + 1],   w2 = w[3 * n + 2];
    size_t o = (size_t)n * COUT + c;
    float v = fmaxf(out[o] * ss[c] + ss[256 + c], 0.f);
    v += w0 * fsub[(size_t)i0 * COUT + c]
       + w1 * fsub[(size_t)i1 * COUT + c]
       + w2 * fsub[(size_t)i2 * COUT + c];
    out[o] = v;
}

// ---------------------------------------------------------------------------
extern "C" void kernel_launch(void* const* d_in, const int* in_sizes, int n_in,
                              void* d_out, int out_size, void* d_ws, size_t ws_size,
                              hipStream_t stream) {
    (void)in_sizes; (void)n_in; (void)out_size; (void)ws_size;
    const float* pos     = (const float*)d_in[0];
    const float* x       = (const float*)d_in[1];
    const float* pos_sub = (const float*)d_in[2];
    const float* x_sub   = (const float*)d_in[3];
    const float* W2      = (const float*)d_in[4];
    const float* b2      = (const float*)d_in[5];
    const float* g2      = (const float*)d_in[6];
    const float* be2     = (const float*)d_in[7];
    const float* W1      = (const float*)d_in[8];
    const float* b1      = (const float*)d_in[9];
    const float* g1      = (const float*)d_in[10];
    const float* be1     = (const float*)d_in[11];
    float* out = (float*)d_out;

    // Workspace carve-up (~26.5 MB, 256B aligned slices)
    char*  ws  = (char*)d_ws;
    size_t off = 0;
    auto alloc = [&](size_t bytes) -> void* {
        void* p = ws + off;
        off += (bytes + 255) & ~(size_t)255;
        return p;
    };
    float*     h2    = (float*)    alloc((size_t)M_SUB * COUT * 4); // h2 -> f_sub (in-place)
    _Float16*  Axs   = (_Float16*) alloc((size_t)M_SUB * CIN  * 2);
    _Float16*  Bw2   = (_Float16*) alloc((size_t)CIN   * COUT * 2);
    _Float16*  Ax    = (_Float16*) alloc((size_t)N_PTS * COUT * 2);
    _Float16*  Bw1   = (_Float16*) alloc((size_t)COUT  * COUT * 2);
    float*     part2 = (float*)    alloc(48 * 512 * 4);
    float*     part1 = (float*)    alloc(96 * 512 * 4);
    float*     ss2   = (float*)    alloc(512 * 4);
    float*     ss1   = (float*)    alloc(512 * 4);
    int*       kidx  = (int*)      alloc((size_t)N_PTS * 3 * 4);
    float*     kw    = (float*)    alloc((size_t)N_PTS * 3 * 4);

    // 1) Pack operands into WMMA f16 layouts
    pack_a_kernel<<<(M_SUB * CIN)  / 256, 256, 0, stream>>>(x_sub, Axs, CIN);
    pack_b_kernel<<<(CIN * COUT)   / 256, 256, 0, stream>>>(W2, Bw2, COUT);
    pack_a_kernel<<<(N_PTS * COUT) / 256, 256, 0, stream>>>(x, Ax, COUT);
    pack_b_kernel<<<(COUT * COUT)  / 256, 256, 0, stream>>>(W1, Bw1, COUT);

    // 2) h2 = x_sub @ W2 + b2   (192 x 8 32x32-wave-tiles, KB=16)
    gemm_f16_kernel<<<192, 256, 0, stream>>>(Axs, Bw2, b2, h2, 192, 8, 16, COUT);

    // 3) BN2 stats + apply -> f_sub (in place)
    bn_partial_kernel<<<48, 256, 0, stream>>>(h2, part2, 128);
    bn_finalize_kernel<<<1, 256, 0, stream>>>(part2, 48, 1.f / M_SUB, g2, be2, ss2);
    bn_relu_kernel<<<(M_SUB * COUT) / 256, 256, 0, stream>>>(h2, ss2);

    // 4) 3-NN with whole pos_sub in LDS
    knn_kernel<<<N_PTS / 256, 256, 0, stream>>>(pos, pos_sub, kidx, kw);

    // 5) h1 = x @ W1 + b1 -> d_out   (768 x 8 32x32-wave-tiles, KB=8)
    gemm_f16_kernel<<<768, 256, 0, stream>>>(Ax, Bw1, b1, out, 768, 8, 8, COUT);

    // 6) BN1 stats + fused BN/ReLU/interp epilogue (in place on d_out)
    bn_partial_kernel<<<96, 256, 0, stream>>>(out, part1, 256);
    bn_finalize_kernel<<<1, 256, 0, stream>>>(part1, 96, 1.f / N_PTS, g1, be1, ss1);
    final_kernel<<<N_PTS, 256, 0, stream>>>(out, h2, ss1, kidx, kw);
}
